// RelationalGraphConvSparse_78769700208705
// MI455X (gfx1250) — matrix operand order repaired and checked by reading, not compiled, not run on hardware
//
#include <hip/hip_runtime.h>
#include <hip/hip_bf16.h>

typedef __attribute__((ext_vector_type(16))) __bf16 v16bf;
typedef __attribute__((ext_vector_type(8)))  float  v8f;

// ---------------------------------------------------------------------------
// GEMM: C[M x 128] = A[M x 128] * W[128 x 128]  (f32 in/out, bf16-split WMMA)
// Block = 256 threads (8 waves). Block owns one 16-row M tile; wave w owns
// the 16-col N tile starting at w*16. K loop in steps of 32, 3 WMMAs per step
// (hi*hi + lo*hi + hi*lo) for ~f32 accuracy.
// ---------------------------------------------------------------------------
__global__ __launch_bounds__(256)
void gemm_rgcn(const float* __restrict__ A, const float* __restrict__ W,
               float* __restrict__ C, int M) {
  const int lane  = threadIdx.x & 31;
  const int wave  = threadIdx.x >> 5;          // n-tile id, 0..7
  const int mbase = blockIdx.x * 16;
  const int nbase = wave * 16;
  const int l16   = lane & 15;
  const int mrow  = mbase + l16;               // A row this lane feeds
  const int ncol  = nbase + l16;               // B/C column this lane feeds
  const int khalfA = (lane >> 4) * 16;         // A: hi lanes hold K=16..31
  const int khalfB = (lane >> 4) * 16;         // B: hi lanes hold K=16..31

  const float* __restrict__ arow = A + (size_t)mrow * 128;
  v8f acc = {};

  for (int k0 = 0; k0 < 128; k0 += 32) {
    v16bf a_hi, a_lo, b_hi, b_lo;
#pragma unroll
    for (int v = 0; v < 8; ++v) {
      // ---- A fragment (16x32 bf16, contiguous K: k = 16*half + s) ----
      const int ka = k0 + khalfA + 2 * v;
      const float x0 = arow[ka];
      const float x1 = arow[ka + 1];
      const __bf16 h0 = (__bf16)x0;
      const __bf16 h1 = (__bf16)x1;
      a_hi[2 * v]     = h0;
      a_hi[2 * v + 1] = h1;
      a_lo[2 * v]     = (__bf16)(x0 - (float)h0);
      a_lo[2 * v + 1] = (__bf16)(x1 - (float)h1);
      // ---- B fragment (32x16 bf16, contiguous K: k = 16*half + s, same as A) ----
      const int kb = k0 + khalfB + 2 * v;
      const float w0 = W[(size_t)kb * 128 + ncol];
      const float w1 = W[(size_t)(kb + 1) * 128 + ncol];
      const __bf16 g0 = (__bf16)w0;
      const __bf16 g1 = (__bf16)w1;
      b_hi[2 * v]     = g0;
      b_hi[2 * v + 1] = g1;
      b_lo[2 * v]     = (__bf16)(w0 - (float)g0);
      b_lo[2 * v + 1] = (__bf16)(w1 - (float)g1);
    }
    acc = __builtin_amdgcn_wmma_f32_16x16x32_bf16(false, a_hi, false, b_hi,
                                                  (short)0, acc, false, false);
    acc = __builtin_amdgcn_wmma_f32_16x16x32_bf16(false, a_lo, false, b_hi,
                                                  (short)0, acc, false, false);
    acc = __builtin_amdgcn_wmma_f32_16x16x32_bf16(false, a_hi, false, b_lo,
                                                  (short)0, acc, false, false);
  }

  // C/D layout: VGPR i -> row i (lanes 0-15) or i+8 (lanes 16-31), col = l16
  const int rbase = mbase + ((lane >> 4) ? 8 : 0);
#pragma unroll
  for (int i = 0; i < 8; ++i) {
    C[(size_t)(rbase + i) * 128 + ncol] = acc[i];
  }
}

// ---------------------------------------------------------------------------
// Edge scatter: one wave per edge (both batches), lane owns one float4 of D.
// out[b, dst, :] += ew[e] * H[b, src, :]   via native f32 global atomics.
// ---------------------------------------------------------------------------
__global__ __launch_bounds__(256)
void scatter_rgcn(const float* __restrict__ H, const float* __restrict__ ew,
                  const int* __restrict__ ei, float* __restrict__ out,
                  int E, int N) {
  const int lane   = threadIdx.x & 31;
  const int gw     = blockIdx.x * (blockDim.x >> 5) + (threadIdx.x >> 5);
  const int stride = gridDim.x * (blockDim.x >> 5);
  const int* __restrict__ srcI = ei;
  const int* __restrict__ dstI = ei + E;
  const size_t bstride = (size_t)N * 128;

  for (int e = gw; e < E; e += stride) {
    const float  w    = ew[e];
    const size_t soff = (size_t)srcI[e] * 128 + (size_t)lane * 4;
    const size_t doff = (size_t)dstI[e] * 128 + (size_t)lane * 4;
#pragma unroll
    for (int b = 0; b < 2; ++b) {
      const float4 v = *(const float4*)(H + b * bstride + soff);
      float* o = out + b * bstride + doff;
      unsafeAtomicAdd(o + 0, w * v.x);
      unsafeAtomicAdd(o + 1, w * v.y);
      unsafeAtomicAdd(o + 2, w * v.z);
      unsafeAtomicAdd(o + 3, w * v.w);
    }
  }
}

// ---------------------------------------------------------------------------
// LayerNorm(eps=1e-3) + ReLU, in place. One wave per row of 128, wave32
// shuffle reductions.
// ---------------------------------------------------------------------------
__global__ __launch_bounds__(256)
void ln_relu_rgcn(float* __restrict__ out, const float* __restrict__ gamma,
                  const float* __restrict__ beta, int Mrows) {
  const int lane = threadIdx.x & 31;
  const int row  = blockIdx.x * (blockDim.x >> 5) + (threadIdx.x >> 5);
  if (row >= Mrows) return;

  float4 x = ((const float4*)(out + (size_t)row * 128))[lane];
  float s = (x.x + x.y) + (x.z + x.w);
#pragma unroll
  for (int off = 16; off > 0; off >>= 1) s += __shfl_xor(s, off, 32);
  const float mu = s * (1.0f / 128.0f);

  const float d0 = x.x - mu, d1 = x.y - mu, d2 = x.z - mu, d3 = x.w - mu;
  float q = (d0 * d0 + d1 * d1) + (d2 * d2 + d3 * d3);
#pragma unroll
  for (int off = 16; off > 0; off >>= 1) q += __shfl_xor(q, off, 32);
  const float rs = rsqrtf(q * (1.0f / 128.0f) + 1e-3f);

  const float4 g = ((const float4*)gamma)[lane];
  const float4 b = ((const float4*)beta)[lane];
  float4 y;
  y.x = fmaxf(d0 * rs * g.x + b.x, 0.0f);
  y.y = fmaxf(d1 * rs * g.y + b.y, 0.0f);
  y.z = fmaxf(d2 * rs * g.z + b.z, 0.0f);
  y.w = fmaxf(d3 * rs * g.w + b.w, 0.0f);
  ((float4*)(out + (size_t)row * 128))[lane] = y;
}

// ---------------------------------------------------------------------------
// Input order: 0 features, 1 self_proj, 2 gamma, 3 beta,
//              4 kernel_r0, 5 edge_weight_r0, 6 edge_index_r0,
//              7 kernel_r1, 8 edge_weight_r1, 9 edge_index_r1,
//              10 kernel_r2, 11 edge_weight_r2, 12 edge_index_r2
// ---------------------------------------------------------------------------
extern "C" void kernel_launch(void* const* d_in, const int* in_sizes, int n_in,
                              void* d_out, int out_size, void* d_ws, size_t ws_size,
                              hipStream_t stream) {
  const float* features  = (const float*)d_in[0];
  const float* self_proj = (const float*)d_in[1];
  const float* gamma     = (const float*)d_in[2];
  const float* beta      = (const float*)d_in[3];
  const float* kernels[3]  = {(const float*)d_in[4], (const float*)d_in[7], (const float*)d_in[10]};
  const float* eweights[3] = {(const float*)d_in[5], (const float*)d_in[8], (const float*)d_in[11]};
  const int*   eindex[3]   = {(const int*)d_in[6], (const int*)d_in[9], (const int*)d_in[12]};

  float* out = (float*)d_out;
  float* h   = (float*)d_ws;   // one relation's h: M*128 floats (51.2 MB)

  const int M = in_sizes[0] / 128;   // B*N = 100000 rows
  const int N = M / 2;               // nodes per batch (B == 2)

  // Self connection overwrites d_out (no memset needed; deterministic replay).
  gemm_rgcn<<<M / 16, 256, 0, stream>>>(features, self_proj, out, M);

  for (int r = 0; r < 3; ++r) {
    const int E = in_sizes[5 + 3 * r];
    gemm_rgcn<<<M / 16, 256, 0, stream>>>(features, kernels[r], h, M);
    scatter_rgcn<<<(E + 7) / 8, 256, 0, stream>>>(h, eweights[r], eindex[r], out, E, N);
  }

  ln_relu_rgcn<<<(M + 7) / 8, 256, 0, stream>>>(out, gamma, beta, M);
}
